// SelfAttention_3770981286069
// MI455X (gfx1250) — matrix operand order repaired and checked by reading, not compiled, not run on hardware
//
#include <hip/hip_runtime.h>
#include <hip/hip_bf16.h>

// Self-attention forward for MI455X (gfx1250), WMMA f16 path.
// B=4, S=2048, D=1024, H=16, HD=64.

#define B_  4
#define S_  2048
#define D_  1024
#define H_  16
#define HD_ 64

typedef _Float16 v16h  __attribute__((ext_vector_type(16)));
typedef _Float16 half8 __attribute__((ext_vector_type(8)));
typedef float    v8f   __attribute__((ext_vector_type(8)));

// ---------------------------------------------------------------------------
// Fragment loaders (wave32), f16 sources only (hot loops are cvt-free).
//
// A-matrix 16x32 f16 (ISA 7.12.2): lane L<16 holds row M=L, K = {kb..kb+7,
// 16+kb..16+kb+7} with kb=0; lanes 16..31: same rows, kb=8.
// B-matrix 32x16 f16: lane L<16 holds column N=L, K = {0..7, 16..23};
// lanes 16..31 hold K = {8..15, 24..31}.  For C = A * W^T with row-major W, column n
// of B is row n of W, so both operands load contiguous 16B-aligned rows.
// Works identically for global and LDS source pointers.
// ---------------------------------------------------------------------------

__device__ __forceinline__ v16h frag_a_f16(const _Float16* rowptr) {
    const int lane = threadIdx.x & 31;
    const int kb = (lane < 16) ? 0 : 8;
    half8 a = *reinterpret_cast<const half8*>(rowptr + kb);
    half8 b = *reinterpret_cast<const half8*>(rowptr + kb + 16);
    v16h f;
#pragma unroll
    for (int j = 0; j < 8; ++j) { f[j] = a[j]; f[8 + j] = b[j]; }
    return f;
}

__device__ __forceinline__ v16h frag_b_f16(const _Float16* rowptr) {
    const int lane = threadIdx.x & 31;
    const int ko = (lane < 16) ? 0 : 8;
    half8 a = *reinterpret_cast<const half8*>(rowptr + ko);
    half8 b = *reinterpret_cast<const half8*>(rowptr + ko + 16);
    v16h f;
#pragma unroll
    for (int j = 0; j < 8; ++j) { f[j] = a[j]; f[8 + j] = b[j]; }
    return f;
}

__device__ __forceinline__ v8f wmma_f16(v16h a, v16h b, v8f c) {
    return __builtin_amdgcn_wmma_f32_16x16x32_f16(
        /*neg_a=*/false, a, /*neg_b=*/false, b,
        /*c_mod=*/(short)0, c, /*reuse_a=*/false, /*reuse_b=*/false);
}

// ---------------------------------------------------------------------------
// Kernel 0: bulk f32 -> f16 convert (x, W_attn, W_out), 8 elems/thread.
// ---------------------------------------------------------------------------
__global__ void __launch_bounds__(256) cvt_kernel(
    const float* __restrict__ in, _Float16* __restrict__ out, int n8)
{
    const int i = blockIdx.x * 256 + threadIdx.x;
    if (i >= n8) return;
    const float4* p = reinterpret_cast<const float4*>(in) + (size_t)i * 2;
    const float4 a = p[0];
    const float4 b = p[1];
    half8 h;
    h[0] = (_Float16)a.x; h[1] = (_Float16)a.y;
    h[2] = (_Float16)a.z; h[3] = (_Float16)a.w;
    h[4] = (_Float16)b.x; h[5] = (_Float16)b.y;
    h[6] = (_Float16)b.z; h[7] = (_Float16)b.w;
    reinterpret_cast<half8*>(out)[i] = h;
}

// ---------------------------------------------------------------------------
// Shared 64x64-per-wave GEMM inner loop: C(64x64) += A(64xK) * B(64xK)^T,
// A_tile/B_tile pre-offset to the tile's first row, ld = K = D_ = 1024.
// 16 WMMAs per 32-wide k-chunk, 16 b128 loads -> 1 load per WMMA.
// ---------------------------------------------------------------------------
__device__ __forceinline__ void gemm64x64(
    const _Float16* __restrict__ A_tile, const _Float16* __restrict__ B_tile,
    v8f acc[4][4])
{
    const int ln = threadIdx.x & 15;
    for (int kc = 0; kc < D_; kc += 32) {
        v16h af[4];
#pragma unroll
        for (int i = 0; i < 4; ++i)
            af[i] = frag_a_f16(A_tile + (size_t)(i * 16 + ln) * D_ + kc);
#pragma unroll
        for (int j = 0; j < 4; ++j) {
            v16h bf = frag_b_f16(B_tile + (size_t)(j * 16 + ln) * D_ + kc);
#pragma unroll
            for (int i = 0; i < 4; ++i)
                acc[i][j] = wmma_f16(af[i], bf, acc[i][j]);
        }
    }
}

// ---------------------------------------------------------------------------
// Kernel 1: fused QKV projection.  kqv = x @ W_attn^T + b_attn, split k|q|v,
// scatter into f16 scratch: Q,K as [B,H,S,HD] (Q pre-scaled by 1/sqrt(HD)),
// V transposed as [B,H,HD,S].  One wave computes a 64(M) x 64(N) tile.
// ---------------------------------------------------------------------------
__global__ void __launch_bounds__(32) qkv_kernel(
    const _Float16* __restrict__ x16, const _Float16* __restrict__ W16,
    const float* __restrict__ bias,
    _Float16* __restrict__ q16, _Float16* __restrict__ k16,
    _Float16* __restrict__ vT16)
{
    const int lane = threadIdx.x & 31;
    const int ln   = lane & 15;
    const int m0   = blockIdx.x * 64;   // token tile over B*S = 8192
    const int n0   = blockIdx.y * 64;   // output-feature tile over 3*D = 3072

    v8f acc[4][4] = {};
    gemm64x64(x16 + (size_t)m0 * D_, W16 + (size_t)n0 * D_, acc);

    const int rowo = (lane < 16) ? 0 : 8;
#pragma unroll
    for (int j = 0; j < 4; ++j) {
        const int e     = n0 + j * 16 + ln;
        const int which = e >> 10;          // 0=k, 1=q, 2=v (k first!)
        const int rr    = e & (D_ - 1);
        const int h     = rr >> 6;
        const int hd    = rr & (HD_ - 1);
        const float bv  = bias[e];
#pragma unroll
        for (int i = 0; i < 4; ++i) {
#pragma unroll
            for (int r = 0; r < 8; ++r) {
                const int sg = m0 + i * 16 + r + rowo;   // global token index
                const int b  = sg >> 11;                 // / S_
                const int s  = sg & (S_ - 1);
                const size_t bh = (size_t)(b * H_ + h);
                const float val = acc[i][j][r] + bv;
                if (which == 0) {
                    k16[(bh * S_ + s) * HD_ + hd] = (_Float16)val;
                } else if (which == 1) {
                    q16[(bh * S_ + s) * HD_ + hd] = (_Float16)(val * 0.125f);
                } else {
                    vT16[(bh * HD_ + hd) * S_ + s] = (_Float16)val;
                }
            }
        }
    }
}

// ---------------------------------------------------------------------------
// Kernel 2: flash attention.  4 waves x 32 queries = 128 queries per block.
// K/V chunks (32 keys) are cooperatively staged into double-buffered LDS
// once per block (4x less L2 traffic than per-wave streaming); next chunk's
// global loads are issued before computing the current one.
// Online softmax with row stats held in C-layout-consistent registers
// (one matrix row == one 16-lane half on wave32, reduced via shfl_xor 1/2/4/8);
// P reshaped C-layout -> A-fragment through per-wave LDS, ordered by a
// wave-local s_wait_dscnt (same-wave DS ops are in-order).
// ---------------------------------------------------------------------------
__global__ void __launch_bounds__(128) attn_kernel(
    const _Float16* __restrict__ q16, const _Float16* __restrict__ k16,
    const _Float16* __restrict__ vT16, _Float16* __restrict__ y16)
{
    __shared__ __align__(16) _Float16 Kt[2][32 * 64];  // [buf][key][hd]
    __shared__ __align__(16) _Float16 Vt[2][64 * 32];  // [buf][hd][key]
    __shared__ __align__(16) _Float16 Ps[4][32 * 32];  // per-wave P tile

    const int tid  = threadIdx.x;
    const int lane = tid & 31;
    const int w    = tid >> 5;          // wave 0..3
    const int ln   = lane & 15;
    const int qt   = blockIdx.x & 15;   // S_/128 = 16 query blocks
    const int bh   = blockIdx.x >> 4;   // 0..63
    const int q0   = qt * 128 + w * 32; // this wave's 32 queries

    const _Float16* qb = q16  + (size_t)bh * S_  * HD_;
    const _Float16* kg = k16  + (size_t)bh * S_  * HD_;
    const _Float16* vg = vT16 + (size_t)bh * HD_ * S_;

    // --- staging assignment: 128 lanes x 16 halves cover each 2048-half tile
    const int krow = tid >> 2, kcol = (tid & 3) * 16;   // K tile: 32 x 64
    const int vrow = tid >> 1, vcol = (tid & 1) * 16;   // V tile: 64 x 32

    half8 kr0, kr1, vr0, vr1;   // in-flight staging registers

    // Q fragments are k-loop stationary: 2 query subtiles x 2 HD chunks.
    v16h aq[2][2];
#pragma unroll
    for (int qa = 0; qa < 2; ++qa) {
        aq[qa][0] = frag_a_f16(qb + (size_t)(q0 + qa * 16 + ln) * HD_ + 0);
        aq[qa][1] = frag_a_f16(qb + (size_t)(q0 + qa * 16 + ln) * HD_ + 32);
    }

    // --- prologue: stage chunk 0 into buffer 0
    {
        const _Float16* kp = kg + (size_t)krow * HD_ + kcol;
        kr0 = *reinterpret_cast<const half8*>(kp);
        kr1 = *reinterpret_cast<const half8*>(kp + 8);
        const _Float16* vp = vg + (size_t)vrow * S_ + vcol;
        vr0 = *reinterpret_cast<const half8*>(vp);
        vr1 = *reinterpret_cast<const half8*>(vp + 8);
        half8* kd = reinterpret_cast<half8*>(&Kt[0][krow * 64 + kcol]);
        kd[0] = kr0; kd[1] = kr1;
        half8* vd = reinterpret_cast<half8*>(&Vt[0][vrow * 32 + vcol]);
        vd[0] = vr0; vd[1] = vr1;
    }
    __syncthreads();

    v8f   yacc[2][4] = {};
    float m[2][8], l[2][8];
#pragma unroll
    for (int qa = 0; qa < 2; ++qa)
#pragma unroll
        for (int r = 0; r < 8; ++r) { m[qa][r] = -1.0e30f; l[qa][r] = 0.0f; }

    const int rowo = (lane < 16) ? 0 : 8;
    int p = 0;

    for (int kc = 0; kc < S_; kc += 32) {
        const bool hasNext = (kc + 32) < S_;
        if (hasNext) {  // issue next chunk's global loads early (latency hide)
            const _Float16* kp = kg + (size_t)(kc + 32 + krow) * HD_ + kcol;
            kr0 = *reinterpret_cast<const half8*>(kp);
            kr1 = *reinterpret_cast<const half8*>(kp + 8);
            const _Float16* vp = vg + (size_t)vrow * S_ + (kc + 32) + vcol;
            vr0 = *reinterpret_cast<const half8*>(vp);
            vr1 = *reinterpret_cast<const half8*>(vp + 8);
        }

        // ---- scores: 2 q-subtiles x 2 k-subtiles (16x16 f32 each) ----
        v8f s[2][2];
#pragma unroll
        for (int kt = 0; kt < 2; ++kt) {
            v16h kf0 = frag_b_f16(&Kt[p][(kt * 16 + ln) * 64 + 0]);
            v16h kf1 = frag_b_f16(&Kt[p][(kt * 16 + ln) * 64 + 32]);
#pragma unroll
            for (int qa = 0; qa < 2; ++qa) {
                v8f z = {};
                z = wmma_f16(aq[qa][0], kf0, z);
                s[qa][kt] = wmma_f16(aq[qa][1], kf1, z);
            }
        }

        // ---- online softmax (per q-subtile, rows live in 16-lane halves) --
#pragma unroll
        for (int qa = 0; qa < 2; ++qa) {
            float scal[8];
#pragma unroll
            for (int r = 0; r < 8; ++r) {
                float t = fmaxf(s[qa][0][r], s[qa][1][r]);
                t = fmaxf(t, __shfl_xor(t, 1, 32));
                t = fmaxf(t, __shfl_xor(t, 2, 32));
                t = fmaxf(t, __shfl_xor(t, 4, 32));
                t = fmaxf(t, __shfl_xor(t, 8, 32));
                const float mnew = fmaxf(m[qa][r], t);
                scal[r]  = __expf(m[qa][r] - mnew);
                m[qa][r] = mnew;
            }
#pragma unroll
            for (int r = 0; r < 8; ++r) {
                const float p0 = __expf(s[qa][0][r] - m[qa][r]);
                const float p1 = __expf(s[qa][1][r] - m[qa][r]);
                const int row = qa * 16 + r + rowo;
                Ps[w][row * 32 + ln]      = (_Float16)p0;
                Ps[w][row * 32 + 16 + ln] = (_Float16)p1;
                float rs = p0 + p1;
                rs += __shfl_xor(rs, 1, 32);
                rs += __shfl_xor(rs, 2, 32);
                rs += __shfl_xor(rs, 4, 32);
                rs += __shfl_xor(rs, 8, 32);
                l[qa][r] = l[qa][r] * scal[r] + rs;
#pragma unroll
                for (int t4 = 0; t4 < 4; ++t4) yacc[qa][t4][r] *= scal[r];
            }
        }

        // Wave-local DS ordering for the Ps write -> read round trip.
        asm volatile("s_wait_dscnt 0x0" ::: "memory");

        // ---- y += P @ V : A = P (16x32 per subtile), B = V^T rows (LDS) ----
        v16h pa[2];
#pragma unroll
        for (int qa = 0; qa < 2; ++qa)
            pa[qa] = frag_a_f16(&Ps[w][(qa * 16 + ln) * 32]);
#pragma unroll
        for (int t4 = 0; t4 < 4; ++t4) {
            v16h bvf = frag_b_f16(&Vt[p][(t4 * 16 + ln) * 32]);
#pragma unroll
            for (int qa = 0; qa < 2; ++qa)
                yacc[qa][t4] = wmma_f16(pa[qa], bvf, yacc[qa][t4]);
        }

        // ---- stage next chunk into the other buffer, then block barrier ----
        if (hasNext) {
            half8* kd = reinterpret_cast<half8*>(&Kt[p ^ 1][krow * 64 + kcol]);
            kd[0] = kr0; kd[1] = kr1;
            half8* vd = reinterpret_cast<half8*>(&Vt[p ^ 1][vrow * 32 + vcol]);
            vd[0] = vr0; vd[1] = vr1;
        }
        __syncthreads();
        p ^= 1;
    }

    // ---- normalize, write y back in [B, S, H*HD] = [B,S,D] f16 layout ----
    const int b = bh >> 4, h = bh & (H_ - 1);
#pragma unroll
    for (int qa = 0; qa < 2; ++qa) {
#pragma unroll
        for (int t4 = 0; t4 < 4; ++t4) {
#pragma unroll
            for (int r = 0; r < 8; ++r) {
                const float val = yacc[qa][t4][r] / l[qa][r];
                const int s = q0 + qa * 16 + r + rowo;
                y16[(size_t)(b * S_ + s) * D_ + h * HD_ + t4 * 16 + ln] =
                    (_Float16)val;
            }
        }
    }
}

// ---------------------------------------------------------------------------
// Kernel 3: output projection. out = y @ W_out^T + b_out (f32 out),
// one wave per 64x64 tile.
// ---------------------------------------------------------------------------
__global__ void __launch_bounds__(32) proj_kernel(
    const _Float16* __restrict__ y16, const _Float16* __restrict__ W16,
    const float* __restrict__ bias, float* __restrict__ out)
{
    const int lane = threadIdx.x & 31;
    const int ln   = lane & 15;
    const int m0   = blockIdx.x * 64;
    const int n0   = blockIdx.y * 64;

    v8f acc[4][4] = {};
    gemm64x64(y16 + (size_t)m0 * D_, W16 + (size_t)n0 * D_, acc);

    const int rowo = (lane < 16) ? 0 : 8;
#pragma unroll
    for (int j = 0; j < 4; ++j) {
        const int e    = n0 + j * 16 + ln;
        const float bv = bias[e];
#pragma unroll
        for (int i = 0; i < 4; ++i) {
#pragma unroll
            for (int r = 0; r < 8; ++r) {
                out[(size_t)(m0 + i * 16 + r + rowo) * D_ + e] =
                    acc[i][j][r] + bv;
            }
        }
    }
}

// ---------------------------------------------------------------------------
extern "C" void kernel_launch(void* const* d_in, const int* in_sizes, int n_in,
                              void* d_out, int out_size, void* d_ws, size_t ws_size,
                              hipStream_t stream) {
    const float* x      = (const float*)d_in[0];
    const float* W_attn = (const float*)d_in[1];
    const float* b_attn = (const float*)d_in[2];
    const float* W_out  = (const float*)d_in[3];
    const float* b_out  = (const float*)d_in[4];
    float* out = (float*)d_out;

    // Scratch layout (f16): x | W_attn | W_out | Q | K | V^T | Y
    const size_t Nx  = (size_t)B_ * S_ * D_;        // 8,388,608
    const size_t Nwa = (size_t)3 * D_ * D_;         // 3,145,728
    const size_t Nwo = (size_t)D_ * D_;             // 1,048,576
    const size_t N1  = (size_t)B_ * H_ * S_ * HD_;  // 8,388,608

    _Float16* x16  = (_Float16*)d_ws;
    _Float16* wa16 = x16  + Nx;
    _Float16* wo16 = wa16 + Nwa;
    _Float16* q16  = wo16 + Nwo;
    _Float16* k16  = q16  + N1;
    _Float16* vT16 = k16  + N1;
    _Float16* y16  = vT16 + N1;

    // 0) one-pass f32 -> f16 conversion (removes all cvt from GEMM hot loops)
    cvt_kernel<<<(int)((Nx  / 8 + 255) / 256), 256, 0, stream>>>(x,      x16,  (int)(Nx  / 8));
    cvt_kernel<<<(int)((Nwa / 8 + 255) / 256), 256, 0, stream>>>(W_attn, wa16, (int)(Nwa / 8));
    cvt_kernel<<<(int)((Nwo / 8 + 255) / 256), 256, 0, stream>>>(W_out,  wo16, (int)(Nwo / 8));

    // 1) QKV projection: M = B*S = 8192, N = 3*D = 3072, 64x64 tiles
    qkv_kernel<<<dim3((B_ * S_) / 64, (3 * D_) / 64), 32, 0, stream>>>(
        x16, wa16, b_attn, q16, k16, vT16);

    // 2) Flash attention: 1024 blocks of 4 waves (128 queries each)
    attn_kernel<<<(B_ * H_ * S_) / 128, 128, 0, stream>>>(q16, k16, vT16, y16);

    // 3) Output projection: M = 8192, N = D = 1024, 64x64 tiles
    proj_kernel<<<dim3((B_ * S_) / 64, D_ / 64), 32, 0, stream>>>(
        y16, wo16, b_out, out);
}
